// MultiHeadAttention_39178691674252
// MI455X (gfx1250) — matrix operand order repaired and checked by reading, not compiled, not run on hardware
//
#include <hip/hip_runtime.h>
#include <stdint.h>

// ---------------------------------------------------------------------------
// Causal single-head attention, B=4, S=2048, D=1024, fp32 in/out.
// Phase A: QKV projection GEMM (fp32 -> bf16 WMMA -> bf16 q/k/v in workspace)
// Phase B: flash attention with online softmax, bf16 WMMA, fp32 output.
// Target: gfx1250 (CDNA5, wave32, v_wmma_f32_16x16x32_bf16,
//         global_load_async_to_lds_b128 + s_wait_asynccnt when available)
// ---------------------------------------------------------------------------

typedef __attribute__((ext_vector_type(16))) __bf16 v16bf;
typedef __attribute__((ext_vector_type(8)))  float  v8f;
typedef __attribute__((ext_vector_type(4)))  int    v4i;

#define D_MODEL 1024
#define SEQ     2048
#define NBATCH  4

// CDNA5 async global->LDS staging (ASYNCcnt-tracked DMA, no VGPR round-trip).
// Builtin prototype (from hipcc diagnostic): (v4i AS1*, v4i AS3*, imm, imm).
#if __has_builtin(__builtin_amdgcn_global_load_async_to_lds_b128) && \
    __has_builtin(__builtin_amdgcn_s_wait_asynccnt)
#define USE_ASYNC_LDS 1
typedef __attribute__((address_space(1))) v4i* as1_v4i_p;
typedef __attribute__((address_space(3))) v4i* as3_v4i_p;
#else
#define USE_ASYNC_LDS 0
#endif

__device__ __forceinline__ unsigned short f2bf(float f) {
  // round-to-nearest-even fp32 -> bf16
  unsigned u = __float_as_uint(f);
  u += 0x7FFFu + ((u >> 16) & 1u);
  return (unsigned short)(u >> 16);
}

union FragB16 {          // one WMMA 16-bit A or B operand (16 x bf16)
  v16bf          v;
  unsigned       u32[8];
  uint4          u128[2];
  unsigned short u16[16];
};

__device__ __forceinline__ v8f v8f_zero() {
  v8f z = {0.f, 0.f, 0.f, 0.f, 0.f, 0.f, 0.f, 0.f};
  return z;
}

// 16-byte global -> LDS stage (async DMA when the toolchain exposes it).
__device__ __forceinline__ void stage16(const unsigned short* __restrict__ g,
                                        unsigned short* l) {
#if USE_ASYNC_LDS
  __builtin_amdgcn_global_load_async_to_lds_b128(
      (as1_v4i_p)(v4i*)(void*)g, (as3_v4i_p)(v4i*)(void*)l, 0, 0);
#else
  *(uint4*)l = *(const uint4*)g;
#endif
}

__device__ __forceinline__ void stage_fence() {
#if USE_ASYNC_LDS
  __builtin_amdgcn_s_wait_asynccnt(0);
#endif
}

// ---------------------------------------------------------------------------
// Phase A: out[m,n] = sum_k X[m,k] * W[n,k]   (Linear with weight [out,in])
// Tile: 128(M) x 128(N) x 32(K). 8 waves, each wave computes 64x32.
// z = 0/1/2 selects Wq/Wk/Wv; q is pre-scaled by 1/32.
// ---------------------------------------------------------------------------
__global__ __launch_bounds__(256) void qkv_proj_kernel(
    const float* __restrict__ X,
    const float* __restrict__ Wq,
    const float* __restrict__ Wk,
    const float* __restrict__ Wv,
    unsigned short* __restrict__ Qb,
    unsigned short* __restrict__ Kb,
    unsigned short* __restrict__ Vb)
{
  constexpr int TK  = 32;
  constexpr int LDA = TK + 8;          // padded row stride (bank-conflict relief)

  __shared__ unsigned short As[128 * LDA];
  __shared__ unsigned short Bs[128 * LDA];

  const int z = blockIdx.z;
  const float* __restrict__ W = (z == 0) ? Wq : (z == 1) ? Wk : Wv;
  unsigned short* __restrict__ Out = (z == 0) ? Qb : (z == 1) ? Kb : Vb;
  const float alpha = (z == 0) ? 0.03125f : 1.0f;   // 1/sqrt(1024) folded into q

  const int tid  = threadIdx.x;
  const int lane = tid & 31;
  const int wv   = tid >> 5;
  const int wm   = wv >> 2;            // 0..1  (64 rows each)
  const int wn   = wv & 3;             // 0..3  (32 cols each)
  const int lr   = lane & 15;
  const int hi   = lane >> 4;

  const int m0 = blockIdx.y * 128;
  const int n0 = blockIdx.x * 128;

  const int srow  = tid >> 1;          // 0..127 staging row
  const int shalf = (tid & 1) * 16;    // col offset 0 or 16

  v8f acc[4][2];
#pragma unroll
  for (int i = 0; i < 4; ++i)
#pragma unroll
    for (int j = 0; j < 2; ++j)
      acc[i][j] = v8f_zero();

  for (int k0 = 0; k0 < D_MODEL; k0 += TK) {
    // ---- stage A (x) and B (W) tiles, fp32 -> bf16 on the fly ----
    {
      const float4* srcA = (const float4*)&X[(size_t)(m0 + srow) * D_MODEL + k0 + shalf];
      unsigned*     dstA = (unsigned*)&As[srow * LDA + shalf];
      const float4* srcB = (const float4*)&W[(size_t)(n0 + srow) * D_MODEL + k0 + shalf];
      unsigned*     dstB = (unsigned*)&Bs[srow * LDA + shalf];
#pragma unroll
      for (int i = 0; i < 4; ++i) {
        float4 fa = srcA[i];
        dstA[i * 2 + 0] = (unsigned)f2bf(fa.x) | ((unsigned)f2bf(fa.y) << 16);
        dstA[i * 2 + 1] = (unsigned)f2bf(fa.z) | ((unsigned)f2bf(fa.w) << 16);
        float4 fb = srcB[i];
        dstB[i * 2 + 0] = (unsigned)f2bf(fb.x) | ((unsigned)f2bf(fb.y) << 16);
        dstB[i * 2 + 1] = (unsigned)f2bf(fb.z) | ((unsigned)f2bf(fb.w) << 16);
      }
    }
    __syncthreads();

    // ---- fragments + WMMA ----
    FragB16 afr[4], bfr[2];
#pragma unroll
    for (int mt = 0; mt < 4; ++mt) {
      const unsigned short* arow = &As[(wm * 64 + mt * 16 + lr) * LDA];
#pragma unroll
      for (int p = 0; p < 8; ++p) {
        const int k2 = ((p & 4) ? 16 : 0) + hi * 8 + 2 * (p & 3);
        afr[mt].u32[p] = *(const unsigned*)&arow[k2];
      }
    }
#pragma unroll
    for (int nt = 0; nt < 2; ++nt) {
      const unsigned short* brow = &Bs[(wn * 32 + nt * 16 + lr) * LDA + hi * 8];
      bfr[nt].u128[0] = *(const uint4*)&brow[0];
      bfr[nt].u128[1] = *(const uint4*)&brow[16];
    }
#pragma unroll
    for (int mt = 0; mt < 4; ++mt)
#pragma unroll
      for (int nt = 0; nt < 2; ++nt)
        acc[mt][nt] = __builtin_amdgcn_wmma_f32_16x16x32_bf16(
            false, afr[mt].v, false, bfr[nt].v, (short)0, acc[mt][nt], false, false);
    __syncthreads();
  }

  // ---- store bf16 result ----
#pragma unroll
  for (int mt = 0; mt < 4; ++mt) {
#pragma unroll
    for (int nt = 0; nt < 2; ++nt) {
      const int n     = n0 + wn * 32 + nt * 16 + lr;
      const int mbase = m0 + wm * 64 + mt * 16 + hi * 8;
#pragma unroll
      for (int r = 0; r < 8; ++r)
        Out[(size_t)(mbase + r) * D_MODEL + n] = f2bf(acc[mt][nt][r] * alpha);
    }
  }
}

// ---------------------------------------------------------------------------
// Phase B: flash attention. One block = one (batch, 32-query tile).
// 8 waves. Scores: each wave computes one 16x16 S-tile over half of d.
// PV: each wave owns a 128-column slice of the output; O lives in registers.
// Q/K tiles staged via async global->LDS DMA (ASYNCcnt); V transposed in
// flight through VGPRs (overlaps with the async K copy).
// ---------------------------------------------------------------------------
__global__ __launch_bounds__(256) void flash_attn_kernel(
    const unsigned short* __restrict__ Qb,
    const unsigned short* __restrict__ Kb,
    const unsigned short* __restrict__ Vb,
    float* __restrict__ Out)
{
  constexpr int LQ = D_MODEL + 8;   // padded stride for Qs/Ks
  constexpr int LV = 32 + 8;        // padded stride for VTs/Ps

  __shared__ unsigned short Qs[32 * LQ];        // 32 queries x 1024 (bf16)
  __shared__ unsigned short Ks[32 * LQ];        // 32 keys    x 1024 (bf16)
  __shared__ unsigned short VTs[D_MODEL * LV];  // V transposed: [d][key] (bf16)
  __shared__ float          Sp[2][32][32];      // partial scores (two d-halves)
  __shared__ unsigned short Ps[32 * LV];        // P = exp(S - m) (bf16)
  __shared__ float          rowM[32], rowL[32], rowScale[32];

  const int bidx = blockIdx.y;                  // batch
  const int qt   = blockIdx.x;                  // query tile
  const int q0   = qt * 32;
  const size_t base = (size_t)bidx * SEQ * D_MODEL;

  const int tid  = threadIdx.x;
  const int lane = tid & 31;
  const int wv   = tid >> 5;
  const int lr   = lane & 15;
  const int hi   = lane >> 4;
  const int cb   = wv * 128;                    // this wave's output column base

  const int srow = tid >> 3;                    // 0..31 staging row
  const int sseg = (tid & 7) * 128;             // 128 cols per thread

  // ---- stage Q tile (persistent; async, waited inside the kt loop) ----
  {
    const unsigned short* src = &Qb[base + (size_t)(q0 + srow) * D_MODEL + sseg];
    unsigned short*       dst = &Qs[srow * LQ + sseg];
#pragma unroll
    for (int i = 0; i < 16; ++i) stage16(src + i * 8, dst + i * 8);
  }
  if (tid < 32) { rowM[tid] = -1e30f; rowL[tid] = 0.f; }

  v8f acc[2][8];
#pragma unroll
  for (int rt = 0; rt < 2; ++rt)
#pragma unroll
    for (int ct = 0; ct < 8; ++ct)
      acc[rt][ct] = v8f_zero();

  for (int kt = 0; kt <= qt; ++kt) {
    const int k0 = kt * 32;
    __syncthreads();   // previous PV done before restaging K/V

    // ---- stage K (async DMA) and V (transposed through VGPRs) ----
    {
      const unsigned short* srcK = &Kb[base + (size_t)(k0 + srow) * D_MODEL + sseg];
      unsigned short*       dstK = &Ks[srow * LQ + sseg];
#pragma unroll
      for (int i = 0; i < 16; ++i) stage16(srcK + i * 8, dstK + i * 8);

      const uint4* srcV = (const uint4*)&Vb[base + (size_t)(k0 + srow) * D_MODEL + sseg];
#pragma unroll
      for (int i = 0; i < 16; ++i) {
        uint4 vq = srcV[i];
        const unsigned short* e = (const unsigned short*)&vq;
        const int d0 = sseg + i * 8;
#pragma unroll
        for (int j = 0; j < 8; ++j)
          VTs[(d0 + j) * LV + srow] = e[j];
      }
    }
    stage_fence();      // ASYNCcnt -> 0 (Q on first trip, K every trip)
    __syncthreads();

    // ---- scores: wave (mi,ni) tile over d-half dh ----
    {
      const int t  = wv & 3;
      const int mi = t >> 1, ni = t & 1;
      const int dh = wv >> 2;
      v8f sc = v8f_zero();
      const unsigned short* arow = &Qs[(mi * 16 + lr) * LQ];
      const unsigned short* brow = &Ks[(ni * 16 + lr) * LQ];
#pragma unroll
      for (int kc = 0; kc < 16; ++kc) {
        const int dbase = dh * 512 + kc * 32;
        FragB16 a, b;
#pragma unroll
        for (int p = 0; p < 8; ++p) {
          const int k2 = dbase + ((p & 4) ? 16 : 0) + hi * 8 + 2 * (p & 3);
          a.u32[p] = *(const unsigned*)&arow[k2];
        }
        b.u128[0] = *(const uint4*)&brow[dbase + hi * 8];
        b.u128[1] = *(const uint4*)&brow[dbase + hi * 8 + 16];
        sc = __builtin_amdgcn_wmma_f32_16x16x32_bf16(
            false, a.v, false, b.v, (short)0, sc, false, false);
      }
#pragma unroll
      for (int r = 0; r < 8; ++r)
        Sp[dh][mi * 16 + hi * 8 + r][ni * 16 + lr] = sc[r];
    }
    __syncthreads();

    // ---- online softmax (wave 0, one row per lane) ----
    if (tid < 32) {
      const int r  = tid;
      const int qg = q0 + r;
      float sv[32];
      float mx = -1e30f;
#pragma unroll
      for (int c = 0; c < 32; ++c) {
        float s = Sp[0][r][c] + Sp[1][r][c];     // 1/sqrt(d) already folded into q
        if (k0 + c > qg) s = -1e30f;             // causal mask
        sv[c] = s;
        mx = fmaxf(mx, s);
      }
      const float mo = rowM[r];
      const float nm = fmaxf(mo, mx);
      float sum = 0.f;
#pragma unroll
      for (int c = 0; c < 32; ++c) {
        const float p = __expf(sv[c] - nm);
        Ps[r * LV + c] = f2bf(p);
        sum += p;
      }
      const float scl = __expf(mo - nm);
      rowL[r]     = rowL[r] * scl + sum;
      rowM[r]     = nm;
      rowScale[r] = scl;
    }
    __syncthreads();

    // ---- rescale O and accumulate P @ V ----
    {
      float scl[2][8];
#pragma unroll
      for (int rt = 0; rt < 2; ++rt)
#pragma unroll
        for (int r = 0; r < 8; ++r)
          scl[rt][r] = rowScale[rt * 16 + hi * 8 + r];

      FragB16 afr[2];
#pragma unroll
      for (int rt = 0; rt < 2; ++rt) {
        const unsigned short* prow = &Ps[(rt * 16 + lr) * LV];
#pragma unroll
        for (int p = 0; p < 8; ++p) {
          const int k2 = ((p & 4) ? 16 : 0) + hi * 8 + 2 * (p & 3);
          afr[rt].u32[p] = *(const unsigned*)&prow[k2];
        }
      }
#pragma unroll
      for (int ct = 0; ct < 8; ++ct) {
        FragB16 b;
        const unsigned short* vrow = &VTs[(cb + ct * 16 + lr) * LV + hi * 8];
        b.u128[0] = *(const uint4*)&vrow[0];
        b.u128[1] = *(const uint4*)&vrow[16];
#pragma unroll
        for (int rt = 0; rt < 2; ++rt) {
#pragma unroll
          for (int r = 0; r < 8; ++r)
            acc[rt][ct][r] *= scl[rt][r];
          acc[rt][ct] = __builtin_amdgcn_wmma_f32_16x16x32_bf16(
              false, afr[rt].v, false, b.v, (short)0, acc[rt][ct], false, false);
        }
      }
    }
  }

  __syncthreads();
  // ---- finalize: O / l, fp32 store ----
  {
    float linv[2][8];
#pragma unroll
    for (int rt = 0; rt < 2; ++rt)
#pragma unroll
      for (int r = 0; r < 8; ++r)
        linv[rt][r] = 1.0f / rowL[rt * 16 + hi * 8 + r];
#pragma unroll
    for (int rt = 0; rt < 2; ++rt)
#pragma unroll
      for (int ct = 0; ct < 8; ++ct)
#pragma unroll
        for (int r = 0; r < 8; ++r) {
          const int m = q0 + rt * 16 + hi * 8 + r;
          const int n = cb + ct * 16 + lr;
          Out[base + (size_t)m * D_MODEL + n] = acc[rt][ct][r] * linv[rt][r];
        }
  }
}

// ---------------------------------------------------------------------------
extern "C" void kernel_launch(void* const* d_in, const int* in_sizes, int n_in,
                              void* d_out, int out_size, void* d_ws, size_t ws_size,
                              hipStream_t stream) {
  (void)in_sizes; (void)n_in; (void)out_size; (void)ws_size;
  const float* X  = (const float*)d_in[0];
  const float* Wq = (const float*)d_in[1];
  const float* Wk = (const float*)d_in[2];
  const float* Wv = (const float*)d_in[3];
  float* out = (float*)d_out;

  unsigned short* qb = (unsigned short*)d_ws;                       // 16 MB
  unsigned short* kb = qb + (size_t)NBATCH * SEQ * D_MODEL;         // 16 MB
  unsigned short* vb = kb + (size_t)NBATCH * SEQ * D_MODEL;         // 16 MB

  // Phase A: q/k/v projections (z selects Wq/Wk/Wv)
  qkv_proj_kernel<<<dim3(D_MODEL / 128, (NBATCH * SEQ) / 128, 3), 256, 0, stream>>>(
      X, Wq, Wk, Wv, qb, kb, vb);

  // Phase B: causal flash attention
  flash_attn_kernel<<<dim3(SEQ / 32, NBATCH), 256, 0, stream>>>(qb, kb, vb, out);
}
